// DeepLSTM_1365799600435
// MI455X (gfx1250) — matrix operand order repaired and checked by reading, not compiled, not run on hardware
//
#include <hip/hip_runtime.h>
#include <hip/hip_bf16.h>

// ---------------------------------------------------------------------------
// DeepLSTM on MI455X (gfx1250): bf16 WMMA everywhere.
//   B=128, T=1024, IN=128, H=256, M1=M2=512, 4 gates.
//   x-path hoisted out of the recurrence (fully parallel fused 3-layer MLP).
//   h-path: persistent kernel, 3 device-wide barriers per step.
// ---------------------------------------------------------------------------

typedef __attribute__((ext_vector_type(16))) __bf16 v16bf;
typedef __attribute__((ext_vector_type(8)))  float  v8f;

#define BB   128
#define TT   1024
#define INN  128
#define HH   256
#define MM1  512
#define MM2  512
#define NROW (TT * BB)          // 131072
#define SEQ_BLOCKS 32
#define SEQ_WAVES  (SEQ_BLOCKS * 8)   // 256 waves

// ---------------- fragment helpers (ISA 7.12.2 layouts, wave32) ------------

// A-matrix 16x32 bf16 fragment from row-major source (lda in elements).
// lane L: row M = L&15; K-half = L>>4.  VGPR0..3 -> K = 8*half + 0..7,
// VGPR4..7 -> K = 16 + 8*half + 0..7  (two contiguous 8-element runs).
__device__ __forceinline__ v16bf load_a_frag(const __bf16* A, int lda, int lane) {
    int r  = lane & 15;
    int hf = lane >> 4;
    const __bf16* p = A + (size_t)r * lda + 8 * hf;
    v16bf f;
#pragma unroll
    for (int i = 0; i < 8; ++i) f[i] = p[i];
#pragma unroll
    for (int i = 0; i < 8; ++i) f[8 + i] = p[16 + i];
    return f;
}

// B fragments are pre-swizzled at weight-conversion time into
// [kt][nt][lane][16] order -> one contiguous 32-byte load per lane.
__device__ __forceinline__ v16bf load_b_frag(const __bf16* Wswz, int lane) {
    return *(const v16bf*)(Wswz + (size_t)lane * 16);
}

// Store 16x16 f32 accumulator (C/D layout: lane L col=L&15, VGPR r row=8*(L>>4)+r)
// as bf16 row-major with bias (+optional relu).
__device__ __forceinline__ void store_tile_bf16(__bf16* out, int ldo, v8f acc,
                                                const float* bias16, int lane,
                                                bool relu) {
    int n  = lane & 15;
    int hf = lane >> 4;
    float bv = bias16[n];
#pragma unroll
    for (int r = 0; r < 8; ++r) {
        float v = acc[r] + bv;
        if (relu) v = v > 0.f ? v : 0.f;
        out[(size_t)(8 * hf + r) * ldo + n] = (__bf16)v;
    }
}

__device__ __forceinline__ float sigmoidf_(float x) {
    return 1.f / (1.f + __expf(-x));
}

#define WMMA_BF16(a, b, c) \
    __builtin_amdgcn_wmma_f32_16x16x32_bf16(false, (a), false, (b), (short)0, (c), false, false)

// ---------------- device-wide barrier for the persistent kernel ------------
__device__ __forceinline__ void grid_barrier(unsigned* bar, unsigned nblocks) {
    __syncthreads();
    if (threadIdx.x == 0) {
        __threadfence();
        volatile unsigned* vb = (volatile unsigned*)bar;
        unsigned gen = vb[1];
        unsigned old = atomicAdd(&bar[0], 1u);
        if (old == nblocks - 1) {
            vb[0] = 0;
            __threadfence();
            atomicAdd(&bar[1], 1u);
        } else {
            while (vb[1] == gen) { __builtin_amdgcn_s_sleep(2); }
        }
        __threadfence();
    }
    __syncthreads();
}

// ---------------- conversion / init kernels --------------------------------

__global__ __launch_bounds__(256) void cvt_f32_bf16(const float* __restrict__ s,
                                                    __bf16* __restrict__ d, size_t n) {
    size_t i = (size_t)blockIdx.x * blockDim.x + threadIdx.x;
    if (i < n) d[i] = (__bf16)s[i];
}

// Convert f32 weights [G,K,N] -> bf16 in B-fragment-swizzled order
// dst[((g*KT+kt)*NT+nt)*32 + lane][j] = src[g, kt*32 + (j<8 ? 8*(lane>>4)+j : 16+8*(lane>>4)+j-8), nt*16 + (lane&15)]
__global__ __launch_bounds__(256) void cvt_w_swz(const float* __restrict__ src,
                                                 __bf16* __restrict__ dst,
                                                 int G, int K, int N) {
    size_t total = (size_t)G * K * N;
    size_t i = (size_t)blockIdx.x * blockDim.x + threadIdx.x;
    if (i >= total) return;
    int j = (int)(i & 15);
    size_t q = i >> 4;
    int lane = (int)(q & 31); q >>= 5;
    int NT = N >> 4, KT = K >> 5;
    int nt = (int)(q % NT); q /= NT;
    int kt = (int)(q % KT);
    int g  = (int)(q / KT);
    int k = kt * 32 + (j < 8 ? 8 * (lane >> 4) + j : 16 + 8 * (lane >> 4) + (j - 8));
    int n = nt * 16 + (lane & 15);
    dst[i] = (__bf16)src[((size_t)g * K + k) * N + n];
}

__global__ __launch_bounds__(256) void init_state(__bf16* hbf, float* cst, unsigned* bar) {
    int i = blockIdx.x * blockDim.x + threadIdx.x;
    if (i < BB * HH) { hbf[i] = (__bf16)0.f; cst[i] = 0.f; }
    if (i < 2) bar[i] = 0u;
}

// ---------------- x-path: fused 3-layer MLP, 16 rows x 1 gate per block ----
// Rows flattened as n = b*T + t (memory order of x). Output scattered into
// ax[t][g][b][h] (bf16) for direct consumption by the recurrence.
__global__ __launch_bounds__(256) void xpath_fused(
    const __bf16* __restrict__ xb,          // [B*T, IN]
    const __bf16* __restrict__ W1s,         // [4][4][32][32][16]   (K=128,N=512)
    const __bf16* __restrict__ W2s,         // [4][16][32][32][16]  (K=512,N=512)
    const __bf16* __restrict__ W3s,         // [4][16][16][32][16]  (K=512,N=256)
    const float* __restrict__ b1,           // [4,512]
    const float* __restrict__ b2,           // [4,512]
    const float* __restrict__ b3,           // [4,256]
    __bf16* __restrict__ ax)                // [T][4][B][H]
{
    __shared__ __bf16 act1[16 * MM1];
    __shared__ __bf16 act2[16 * MM2];

    const int lane = threadIdx.x & 31;
    const int w    = threadIdx.x >> 5;      // 0..7
    const int row0 = blockIdx.x * 16;
    const int g    = blockIdx.y;

    // ---- layer 1: [16,128] x [128,512] ----
    const __bf16* A0 = xb + (size_t)row0 * INN;
#pragma unroll
    for (int nn = 0; nn < 4; ++nn) {
        int nt = w + nn * 8;                // 0..31
        v8f acc = {};
#pragma unroll
        for (int ks = 0; ks < 4; ++ks) {
            v16bf a = load_a_frag(A0 + ks * 32, INN, lane);
            v16bf b = load_b_frag(W1s + ((((size_t)g * 4 + ks) * 32 + nt) * 32) * 16, lane);
            acc = WMMA_BF16(a, b, acc);
        }
        store_tile_bf16(act1 + nt * 16, MM1, acc, b1 + g * MM1 + nt * 16, lane, true);
    }
    __syncthreads();

    // ---- layer 2: [16,512] x [512,512] ----
#pragma unroll
    for (int nn = 0; nn < 4; ++nn) {
        int nt = w + nn * 8;
        v8f acc = {};
#pragma unroll
        for (int ks = 0; ks < 16; ++ks) {
            v16bf a = load_a_frag(act1 + ks * 32, MM1, lane);
            v16bf b = load_b_frag(W2s + ((((size_t)g * 16 + ks) * 32 + nt) * 32) * 16, lane);
            acc = WMMA_BF16(a, b, acc);
        }
        store_tile_bf16(act2 + nt * 16, MM2, acc, b2 + g * MM2 + nt * 16, lane, true);
    }
    __syncthreads();

    // ---- layer 3: [16,512] x [512,256], scatter to ax[t][g][b][h] ----
#pragma unroll
    for (int nn = 0; nn < 2; ++nn) {
        int nt = w + nn * 8;                // 0..15
        v8f acc = {};
#pragma unroll
        for (int ks = 0; ks < 16; ++ks) {
            v16bf a = load_a_frag(act2 + ks * 32, MM2, lane);
            v16bf b = load_b_frag(W3s + ((((size_t)g * 16 + ks) * 16 + nt) * 32) * 16, lane);
            acc = WMMA_BF16(a, b, acc);
        }
        int nloc = lane & 15, hf = lane >> 4;
        int h = nt * 16 + nloc;
        float bv = b3[g * HH + h];
#pragma unroll
        for (int r = 0; r < 8; ++r) {
            int n  = row0 + 8 * hf + r;     // n = b*T + t
            int bb = n >> 10;               // /T
            int tt = n & 1023;              // %T
            ax[(((size_t)tt * 4 + g) * BB + bb) * HH + h] = (__bf16)(acc[r] + bv);
        }
    }
}

// ---------------- persistent sequential h-path + LSTM cell -----------------
__global__ __launch_bounds__(256) void lstm_seq(
    const __bf16* __restrict__ Wh1s,  // [4][8][32][32][16]   (K=256,N=512)
    const __bf16* __restrict__ Wh2s,  // [4][16][32][32][16]  (K=512,N=512)
    const __bf16* __restrict__ Wh3s,  // [4][16][16][32][16]  (K=512,N=256)
    const float* __restrict__ bh1, const float* __restrict__ bh2,
    const float* __restrict__ bh3,
    const __bf16* __restrict__ ax,    // [T][4][B][H]
    __bf16* __restrict__ a1,          // [4][B][512]
    __bf16* __restrict__ a2,          // [4][B][512]
    __bf16* __restrict__ hbf,         // [B][H]  (A-matrix for next step)
    float*  __restrict__ cst,         // [B][H]
    __bf16* __restrict__ hs,          // [T][B][H]
    unsigned* bar)
{
    const int lane = threadIdx.x & 31;
    const int wave = blockIdx.x * 8 + (threadIdx.x >> 5);   // 0..255

    for (int t = 0; t < TT; ++t) {
        // ===== phase 1: a1 = relu(h * Wh1 + bh1)   [4,128,512], K=256 =====
        for (int idx = wave; idx < 1024; idx += SEQ_WAVES) {
            int g = idx >> 8, rem = idx & 255, bt = rem >> 5, nt = rem & 31;
            const __bf16* A0 = hbf + (size_t)bt * 16 * HH;
            v8f acc = {};
#pragma unroll
            for (int ks = 0; ks < 8; ++ks) {
                v16bf a = load_a_frag(A0 + ks * 32, HH, lane);
                v16bf b = load_b_frag(Wh1s + ((((size_t)g * 8 + ks) * 32 + nt) * 32) * 16, lane);
                acc = WMMA_BF16(a, b, acc);
            }
            store_tile_bf16(a1 + ((size_t)g * BB + bt * 16) * MM1 + nt * 16, MM1,
                            acc, bh1 + g * MM1 + nt * 16, lane, true);
        }
        grid_barrier(bar, SEQ_BLOCKS);

        // ===== phase 2: a2 = relu(a1 * Wh2 + bh2)  [4,128,512], K=512 =====
        for (int idx = wave; idx < 1024; idx += SEQ_WAVES) {
            int g = idx >> 8, rem = idx & 255, bt = rem >> 5, nt = rem & 31;
            const __bf16* A0 = a1 + (size_t)g * BB * MM1 + (size_t)bt * 16 * MM1;
            v8f acc = {};
#pragma unroll
            for (int ks = 0; ks < 16; ++ks) {
                v16bf a = load_a_frag(A0 + ks * 32, MM1, lane);
                v16bf b = load_b_frag(Wh2s + ((((size_t)g * 16 + ks) * 32 + nt) * 32) * 16, lane);
                acc = WMMA_BF16(a, b, acc);
            }
            store_tile_bf16(a2 + ((size_t)g * BB + bt * 16) * MM2 + nt * 16, MM2,
                            acc, bh2 + g * MM2 + nt * 16, lane, true);
        }
        grid_barrier(bar, SEQ_BLOCKS);

        // ===== phase 3+4: gate GEMMs (all 4 gates per wave) + cell update ==
        if (wave < 128) {
            int bt = wave >> 4, ht = wave & 15;
            v8f accg[4];
#pragma unroll
            for (int g = 0; g < 4; ++g) {
                const __bf16* A0 = a2 + (size_t)g * BB * MM2 + (size_t)bt * 16 * MM2;
                v8f acc = {};
#pragma unroll
                for (int ks = 0; ks < 16; ++ks) {
                    v16bf a = load_a_frag(A0 + ks * 32, MM2, lane);
                    v16bf b = load_b_frag(Wh3s + ((((size_t)g * 16 + ks) * 16 + ht) * 32) * 16, lane);
                    acc = WMMA_BF16(a, b, acc);
                }
                accg[g] = acc;
            }
            int nloc = lane & 15, hf = lane >> 4;
            int h = ht * 16 + nloc;
            float bF = bh3[0 * HH + h], bI = bh3[1 * HH + h];
            float bO = bh3[2 * HH + h], bC = bh3[3 * HH + h];
#pragma unroll
            for (int r = 0; r < 8; ++r) {
                int bb = bt * 16 + 8 * hf + r;
                size_t ci  = (size_t)bb * HH + h;
                size_t ab  = (((size_t)t * 4) * BB + bb) * HH + h;   // gate stride B*H
                float aF = accg[0][r] + bF + (float)ax[ab + 0 * (size_t)BB * HH];
                float aI = accg[1][r] + bI + (float)ax[ab + 1 * (size_t)BB * HH];
                float aO = accg[2][r] + bO + (float)ax[ab + 2 * (size_t)BB * HH];
                float aC = accg[3][r] + bC + (float)ax[ab + 3 * (size_t)BB * HH];
                float F = sigmoidf_(aF), I = sigmoidf_(aI), O = sigmoidf_(aO);
                float Ch = tanhf(aC);
                float cv = F * cst[ci] + I * Ch;
                cst[ci] = cv;
                float hv = O * tanhf(cv);
                hbf[ci] = (__bf16)hv;
                hs[(size_t)t * (BB * HH) + ci] = (__bf16)hv;
            }
        }
        grid_barrier(bar, SEQ_BLOCKS);
    }
}

// ---------------- attention: logits = tanh(hs*Wa + ba) ---------------------
__global__ __launch_bounds__(256) void attn_logits(
    const __bf16* __restrict__ hs,    // [T*B, H]
    const __bf16* __restrict__ Was,   // [8][16][32][16] (K=256,N=256)
    const float* __restrict__ ba,     // [H]
    __bf16* __restrict__ logits)      // [T*B, H]
{
    const int lane = threadIdx.x & 31;
    const int w    = threadIdx.x >> 5;
    const int row0 = blockIdx.x * 16;
    const __bf16* A0 = hs + (size_t)row0 * HH;
#pragma unroll
    for (int nn = 0; nn < 2; ++nn) {
        int nt = w + nn * 8;
        v8f acc = {};
#pragma unroll
        for (int ks = 0; ks < 8; ++ks) {
            v16bf a = load_a_frag(A0 + ks * 32, HH, lane);
            v16bf b = load_b_frag(Was + (((size_t)ks * 16 + nt) * 32) * 16, lane);
            acc = WMMA_BF16(a, b, acc);
        }
        int nloc = lane & 15, hf = lane >> 4;
        int h = nt * 16 + nloc;
        float bv = ba[h];
#pragma unroll
        for (int r = 0; r < 8; ++r) {
            float v = tanhf(acc[r] + bv);
            logits[(size_t)(row0 + 8 * hf + r) * HH + h] = (__bf16)v;
        }
    }
}

// ---------------- attention: online softmax over t + weighted sum ----------
__global__ __launch_bounds__(256) void attn_reduce(
    const __bf16* __restrict__ logits,   // [T][B*H]
    const __bf16* __restrict__ hs,       // [T][B*H]
    float* __restrict__ out)             // [B*H]
{
    int i = blockIdx.x * blockDim.x + threadIdx.x;   // b*H + h
    if (i >= BB * HH) return;
    float m = -3.4e38f, l = 0.f;
    for (int t = 0; t < TT; ++t) {
        float s = (float)logits[(size_t)t * (BB * HH) + i];
        if (s > m) { l = l * __expf(m - s); m = s; }
        l += __expf(s - m);
    }
    float inv = 1.f / l;
    float ctx = 0.f;
    for (int t = 0; t < TT; ++t) {
        float s = (float)logits[(size_t)t * (BB * HH) + i];
        float hv = (float)hs[(size_t)t * (BB * HH) + i];
        ctx += __expf(s - m) * inv * hv;
    }
    out[i] = ctx;
}

// ---------------------------------------------------------------------------
extern "C" void kernel_launch(void* const* d_in, const int* in_sizes, int n_in,
                              void* d_out, int out_size, void* d_ws, size_t ws_size,
                              hipStream_t stream) {
    (void)in_sizes; (void)n_in; (void)out_size; (void)ws_size;

    const float* x   = (const float*)d_in[0];
    const float* Wh1 = (const float*)d_in[1];
    const float* bh1 = (const float*)d_in[2];
    const float* Wh2 = (const float*)d_in[3];
    const float* bh2 = (const float*)d_in[4];
    const float* Wh3 = (const float*)d_in[5];
    const float* bh3 = (const float*)d_in[6];
    const float* Wx1 = (const float*)d_in[7];
    const float* bx1 = (const float*)d_in[8];
    const float* Wx2 = (const float*)d_in[9];
    const float* bx2 = (const float*)d_in[10];
    const float* Wx3 = (const float*)d_in[11];
    const float* bx3 = (const float*)d_in[12];
    const float* Wa  = (const float*)d_in[13];
    const float* ba  = (const float*)d_in[14];

    char* ws = (char*)d_ws;
    size_t off = 0;
    auto alloc = [&](size_t bytes) -> void* {
        void* p = ws + off;
        off += (bytes + 255) & ~(size_t)255;
        return p;
    };

    __bf16* xb    = (__bf16*)alloc((size_t)NROW * INN * 2);       // 33.5 MB
    __bf16* Wh1s  = (__bf16*)alloc((size_t)4 * HH  * MM1 * 2);
    __bf16* Wh2s  = (__bf16*)alloc((size_t)4 * MM1 * MM2 * 2);
    __bf16* Wh3s  = (__bf16*)alloc((size_t)4 * MM2 * HH  * 2);
    __bf16* Wx1s  = (__bf16*)alloc((size_t)4 * INN * MM1 * 2);
    __bf16* Wx2s  = (__bf16*)alloc((size_t)4 * MM1 * MM2 * 2);
    __bf16* Wx3s  = (__bf16*)alloc((size_t)4 * MM2 * HH  * 2);
    __bf16* Was   = (__bf16*)alloc((size_t)HH * HH * 2);
    __bf16* axb   = (__bf16*)alloc((size_t)TT * 4 * BB * HH * 2); // 268 MB
    __bf16* hsb   = (__bf16*)alloc((size_t)TT * BB * HH * 2);     // 67 MB
    __bf16* a1    = (__bf16*)alloc((size_t)4 * BB * MM1 * 2);
    __bf16* a2    = (__bf16*)alloc((size_t)4 * BB * MM2 * 2);
    __bf16* hbf   = (__bf16*)alloc((size_t)BB * HH * 2);
    float*  cst   = (float*) alloc((size_t)BB * HH * 4);
    __bf16* logit = (__bf16*)alloc((size_t)TT * BB * HH * 2);     // 67 MB
    unsigned* bar = (unsigned*)alloc(256);

    // 1) convert input + weights (weights into B-fragment-swizzled bf16)
    {
        size_t n = (size_t)NROW * INN;
        cvt_f32_bf16<<<(unsigned)((n + 255) / 256), 256, 0, stream>>>(x, xb, n);
    }
    cvt_w_swz<<<(4 * HH  * MM1 + 255) / 256, 256, 0, stream>>>(Wh1, Wh1s, 4, HH,  MM1);
    cvt_w_swz<<<(4 * MM1 * MM2 + 255) / 256, 256, 0, stream>>>(Wh2, Wh2s, 4, MM1, MM2);
    cvt_w_swz<<<(4 * MM2 * HH  + 255) / 256, 256, 0, stream>>>(Wh3, Wh3s, 4, MM2, HH);
    cvt_w_swz<<<(4 * INN * MM1 + 255) / 256, 256, 0, stream>>>(Wx1, Wx1s, 4, INN, MM1);
    cvt_w_swz<<<(4 * MM1 * MM2 + 255) / 256, 256, 0, stream>>>(Wx2, Wx2s, 4, MM1, MM2);
    cvt_w_swz<<<(4 * MM2 * HH  + 255) / 256, 256, 0, stream>>>(Wx3, Wx3s, 4, MM2, HH);
    cvt_w_swz<<<(HH * HH       + 255) / 256, 256, 0, stream>>>(Wa,  Was,  1, HH,  HH);

    // 2) zero h0/c0 + barrier state (ws is poisoned, must init every call)
    init_state<<<(BB * HH + 255) / 256, 256, 0, stream>>>(hbf, cst, bar);

    // 3) x-path: fully parallel fused 3-layer MLP over all T*B rows, 4 gates
    xpath_fused<<<dim3(NROW / 16, 4), 256, 0, stream>>>(
        xb, Wx1s, Wx2s, Wx3s, bx1, bx2, bx3, axb);

    // 4) sequential recurrence: persistent kernel, 3 grid barriers per step
    lstm_seq<<<SEQ_BLOCKS, 256, 0, stream>>>(
        Wh1s, Wh2s, Wh3s, bh1, bh2, bh3, axb, a1, a2, hbf, cst, hsb, bar);

    // 5) attention
    attn_logits<<<NROW / 16, 256, 0, stream>>>(hsb, Was, ba, logit);
    attn_reduce<<<(BB * HH + 255) / 256, 256, 0, stream>>>(logit, hsb, (float*)d_out);
}
